// TransformerAutoR_71820443124181
// MI455X (gfx1250) — compile-verified, matrix operand order checked by reading, not run on hardware
//
#include <hip/hip_runtime.h>
#include <hip/hip_bf16.h>
#include <cstdint>

// ---------------- problem constants (from reference) ----------------
#define B_SZ     4
#define S_PAST_  1024
#define S_FUT_   512
#define N_TOT    1536                  // S_PAST + S_FUT
#define D_MODEL_ 256
#define N_HEADS_ 8
#define D_HEAD_  32
#define N_LAYERS_ 8
#define D_FF_    2048
#define M_ROWS   (B_SZ * N_TOT)        // 6144
#define CHUNK_   128
#define NCH_     (N_TOT / CHUNK_)      // 12

typedef __attribute__((ext_vector_type(16))) __bf16 v16bf_t;
typedef __attribute__((ext_vector_type(8)))  float  v8f_t;
typedef __attribute__((ext_vector_type(4)))  int    v4i_t;

// CDNA5 async global->LDS copy (ASYNCcnt-tracked), with sync fallback.
#if __has_builtin(__builtin_amdgcn_global_load_async_to_lds_b128) && \
    __has_builtin(__builtin_amdgcn_s_wait_asynccnt)
#define ASYNC_PATH 1
#else
#define ASYNC_PATH 0
#endif

__device__ __forceinline__ void cp16(void* lds, const void* gmem) {
#if ASYNC_PATH
    void* g = const_cast<void*>(gmem);
    __builtin_amdgcn_global_load_async_to_lds_b128(
        (__attribute__((address_space(1))) v4i_t*)g,
        (__attribute__((address_space(3))) v4i_t*)lds, 0, 0);
#else
    *(uint4*)lds = *(const uint4*)gmem;
#endif
}

__device__ __forceinline__ float elu1(float x) {   // elu(x)+1
    return x > 0.f ? x + 1.f : __expf(x);
}

// ---------------- f32[K][N] -> bf16[N][K] tiled transpose (weights, once per call) ------------
__global__ __launch_bounds__(256) void cvt_transpose_kernel(
        const float* __restrict__ src, __bf16* __restrict__ dst,
        int K, int N, long srcLayerStride, long dstLayerStride) {
    __shared__ __bf16 t[32][33];
    const float* S = src + (size_t)blockIdx.z * srcLayerStride;
    __bf16*      D = dst + (size_t)blockIdx.z * dstLayerStride;
    const int k0 = blockIdx.y * 32, n0 = blockIdx.x * 32;
    const int tx = threadIdx.x & 31, ty = threadIdx.x >> 5;   // 32 x 8
    #pragma unroll
    for (int r = 0; r < 32; r += 8)
        t[ty + r][tx] = (__bf16)S[(size_t)(k0 + ty + r) * N + (n0 + tx)];
    __syncthreads();
    #pragma unroll
    for (int r = 0; r < 32; r += 8)
        D[(size_t)(n0 + ty + r) * K + (k0 + tx)] = t[tx][ty + r];
}

// ---------------- concat QKV biases: [L][768] ----------------
__global__ void concat_bias_kernel(const float* __restrict__ bq, const float* __restrict__ bk,
                                   const float* __restrict__ bv, float* __restrict__ bqkv) {
    int i = blockIdx.x * blockDim.x + threadIdx.x;   // L*768
    int l = i / 768, c = i - l * 768;
    float v;
    if (c < 256)      v = bq[l * 256 + c];
    else if (c < 512) v = bk[l * 256 + (c - 256)];
    else              v = bv[l * 256 + (c - 512)];
    bqkv[i] = v;
}

// ---------------- encoder: concat(past/future inputs) @ enc_W + enc_b ----------------
__global__ void encode_kernel(const float* __restrict__ px, const float* __restrict__ py,
                              const float* __restrict__ fx, const float* __restrict__ encW,
                              const float* __restrict__ encb, float* __restrict__ X,
                              __bf16* __restrict__ Xbf) {
    int idx = blockIdx.x * blockDim.x + threadIdx.x;         // M_ROWS * 256
    int c   = idx & 255;
    int row = idx >> 8;
    int b   = row / N_TOT;
    int n   = row - b * N_TOT;
    float in[24];
    if (n < S_PAST_) {
        const float* p16 = px + ((size_t)(b * S_PAST_ + n)) * 16;
        const float* p8  = py + ((size_t)(b * S_PAST_ + n)) * 8;
        #pragma unroll
        for (int k = 0; k < 16; ++k) in[k] = p16[k];
        #pragma unroll
        for (int k = 0; k < 8; ++k)  in[16 + k] = p8[k];
    } else {
        int t = n - S_PAST_;
        const float* p16 = fx + ((size_t)(b * S_FUT_ + t)) * 16;
        const float* p8  = py + ((size_t)(b * S_PAST_ + (S_PAST_ - 1))) * 8;
        #pragma unroll
        for (int k = 0; k < 16; ++k) in[k] = p16[k];
        #pragma unroll
        for (int k = 0; k < 8; ++k)  in[16 + k] = p8[k];
    }
    float acc = encb[c];
    #pragma unroll
    for (int k = 0; k < 24; ++k) acc = fmaf(in[k], encW[k * 256 + c], acc);
    X[idx]   = acc;
    Xbf[idx] = (__bf16)acc;
}

// ---------------- WMMA bf16 GEMM:  C[M,N] = A[M,K](bf16) * Wt[N,K]^T + bias ------------------
// 256 thr = 8 waves; block tile 128x64; wave tile 32x32 (4 WMMA accumulators).
// A and B both bf16, staged with async global->LDS b128 copies, double-buffered on ASYNCcnt.
// Output f32 or bf16 (outBf16), optional ReLU.
__global__ __launch_bounds__(256) void gemm_bf16_kernel(
        const __bf16* __restrict__ A, const __bf16* __restrict__ Wt,
        const float* __restrict__ bias, void* __restrict__ Cout,
        int M, int N, int K, int relu, int outBf16) {
    __shared__ __bf16 Asb[2][128][40];  // 128x32 bf16, row stride 80B (16B aligned, conflict-free)
    __shared__ __bf16 Bsb[2][64][40];   //  64x32 bf16

    const int tid  = threadIdx.x;
    const int lane = tid & 31;
    const int wid  = tid >> 5;
    const int wm   = wid & 3;            // wave row strip -> 32 rows
    const int wn   = wid >> 2;           // wave col strip -> 32 cols
    const int rowBase = blockIdx.y * 128;
    const int colBase = blockIdx.x * 64;
    const int rr = lane & 15;
    const int kg = lane >> 4;            // lane half -> K subgroup (ISA 16-bit frag layout)
    const int nTiles = K >> 5;

    // staging chunk coords (16B = 8 bf16 chunks)
    const int s_r = tid >> 2, s_j = (tid & 3) * 8;   // A: rows s_r, s_r+64 (2/thread); B: 1/thread

    const __bf16* Ag = A  + (size_t)(rowBase + s_r) * K + s_j;
    const __bf16* Bg = Wt + (size_t)(colBase + s_r) * K + s_j;

    v8f_t acc00 = {}, acc01 = {}, acc10 = {}, acc11 = {};

    // prologue: stage tile 0 into buffer 0
    cp16(&Asb[0][s_r][s_j],      Ag);
    cp16(&Asb[0][s_r + 64][s_j], Ag + (size_t)64 * K);
    cp16(&Bsb[0][s_r][s_j],      Bg);

    for (int t = 0; t < nTiles; ++t) {
        const int buf = t & 1;
        if (t + 1 < nTiles) {
            const int k0n = (t + 1) << 5;
            cp16(&Asb[buf ^ 1][s_r][s_j],      Ag + k0n);
            cp16(&Asb[buf ^ 1][s_r + 64][s_j], Ag + (size_t)64 * K + k0n);
            cp16(&Bsb[buf ^ 1][s_r][s_j],      Bg + k0n);
#if ASYNC_PATH
            __builtin_amdgcn_s_wait_asynccnt(3);   // tile t landed; t+1 in flight
#endif
        } else {
#if ASYNC_PATH
            __builtin_amdgcn_s_wait_asynccnt(0);
#endif
        }
        __syncthreads();

        // fragments: per-lane data = two contiguous 16B runs (ISA 7.12.2 16-bit layouts)
        const __bf16 (*As)[40] = Asb[buf];
        const __bf16 (*Bs)[40] = Bsb[buf];
        const int arow0 = wm * 32 + rr;
        const int arow1 = arow0 + 16;
        const int bcol  = wn * 32 + rr;
        const int kb    = kg * 8;
        v16bf_t af0, af1, bv0, bv1;
        #pragma unroll
        for (int e = 0; e < 8; ++e) {
            af0[e]     = As[arow0][kb + e];
            af0[e + 8] = As[arow0][kb + 16 + e];
            af1[e]     = As[arow1][kb + e];
            af1[e + 8] = As[arow1][kb + 16 + e];
            bv0[e]     = Bs[bcol][kb + e];
            bv0[e + 8] = Bs[bcol][kb + 16 + e];
            bv1[e]     = Bs[bcol + 16][kb + e];
            bv1[e + 8] = Bs[bcol + 16][kb + 16 + e];
        }
        acc00 = __builtin_amdgcn_wmma_f32_16x16x32_bf16(false, af0, false, bv0, (short)0, acc00, false, false);
        acc01 = __builtin_amdgcn_wmma_f32_16x16x32_bf16(false, af0, false, bv1, (short)0, acc01, false, false);
        acc10 = __builtin_amdgcn_wmma_f32_16x16x32_bf16(false, af1, false, bv0, (short)0, acc10, false, false);
        acc11 = __builtin_amdgcn_wmma_f32_16x16x32_bf16(false, af1, false, bv1, (short)0, acc11, false, false);
        __syncthreads();
    }

    // epilogue: C layout (ISA C/D table): VGPR p -> row = p + 8*kg, col = lane&15
    const int ccol0 = colBase + wn * 32 + rr;
    const int crow0 = rowBase + wm * 32 + 8 * kg;
    const float b0 = bias[ccol0];
    const float b1 = bias[ccol0 + 16];
    float* Cf = (float*)Cout;
    __bf16* Cb = (__bf16*)Cout;
    #pragma unroll
    for (int p = 0; p < 8; ++p) {
        float v00 = acc00[p] + b0, v01 = acc01[p] + b1;
        float v10 = acc10[p] + b0, v11 = acc11[p] + b1;
        if (relu) {
            v00 = fmaxf(v00, 0.f); v01 = fmaxf(v01, 0.f);
            v10 = fmaxf(v10, 0.f); v11 = fmaxf(v11, 0.f);
        }
        const size_t r0 = (size_t)(crow0 + p) * N;
        const size_t r1 = (size_t)(crow0 + 16 + p) * N;
        if (outBf16) {
            Cb[r0 + ccol0]      = (__bf16)v00;
            Cb[r0 + ccol0 + 16] = (__bf16)v01;
            Cb[r1 + ccol0]      = (__bf16)v10;
            Cb[r1 + ccol0 + 16] = (__bf16)v11;
        } else {
            Cf[r0 + ccol0]      = v00;
            Cf[r0 + ccol0 + 16] = v01;
            Cf[r1 + ccol0]      = v10;
            Cf[r1 + ccol0 + 16] = v11;
        }
    }
}

// ---------------- chunked causal linear attention (3-pass scan) ----------------
// QKV packed [M][768] f32: q at col h*32, k at 256+h*32, v at 512+h*32.
// lane m owns column m of the 32x32 kv state; lane d supplies k/q/v element d via shuffles.

__global__ void attn_partial_kernel(const float* __restrict__ QKV,
                                    float* __restrict__ chS, float* __restrict__ chK) {
    const int lane = threadIdx.x;
    const int blk  = blockIdx.x;
    const int ch   = blk % NCH_;
    const int bh   = blk / NCH_;
    const int b    = bh / N_HEADS_;
    const int h    = bh % N_HEADS_;

    float s[32];
    #pragma unroll
    for (int d = 0; d < 32; ++d) s[d] = 0.f;
    float kc = 0.f;

    for (int n = ch * CHUNK_; n < (ch + 1) * CHUNK_; ++n) {
        size_t base = ((size_t)(b * N_TOT + n)) * 768 + h * D_HEAD_;
        float kd = elu1(QKV[base + 256 + lane]);
        float vd = QKV[base + 512 + lane];
        kc += kd;
        #pragma unroll
        for (int d = 0; d < 32; ++d) {
            float kb = __shfl(kd, d, 32);
            s[d] = fmaf(kb, vd, s[d]);
        }
    }
    #pragma unroll
    for (int d = 0; d < 32; ++d)
        chS[((size_t)(bh * NCH_ + ch) * 32 + d) * 32 + lane] = s[d];
    chK[(size_t)(bh * NCH_ + ch) * 32 + lane] = kc;
}

__global__ void attn_prefix_kernel(float* __restrict__ chS, float* __restrict__ chK) {
    const int lane = threadIdx.x;
    const int bh   = blockIdx.x;
    float runS[32];
    #pragma unroll
    for (int d = 0; d < 32; ++d) runS[d] = 0.f;
    float runK = 0.f;
    for (int ch = 0; ch < NCH_; ++ch) {
        #pragma unroll
        for (int d = 0; d < 32; ++d) {
            size_t idx = ((size_t)(bh * NCH_ + ch) * 32 + d) * 32 + lane;
            float t = chS[idx];
            chS[idx] = runS[d];
            runS[d] += t;
        }
        size_t idk = (size_t)(bh * NCH_ + ch) * 32 + lane;
        float t = chK[idk];
        chK[idk] = runK;
        runK += t;
    }
}

__global__ void attn_out_kernel(const float* __restrict__ QKV, const float* __restrict__ chS,
                                const float* __restrict__ chK, __bf16* __restrict__ O) {
    const int lane = threadIdx.x;
    const int blk  = blockIdx.x;
    const int ch   = blk % NCH_;
    const int bh   = blk / NCH_;
    const int b    = bh / N_HEADS_;
    const int h    = bh % N_HEADS_;

    float s[32];
    #pragma unroll
    for (int d = 0; d < 32; ++d)
        s[d] = chS[((size_t)(bh * NCH_ + ch) * 32 + d) * 32 + lane];
    float kc = chK[(size_t)(bh * NCH_ + ch) * 32 + lane];   // lane acts as d for kcum

    for (int n = ch * CHUNK_; n < (ch + 1) * CHUNK_; ++n) {
        size_t base = ((size_t)(b * N_TOT + n)) * 768 + h * D_HEAD_;
        float qd = elu1(QKV[base + lane]);
        float kd = elu1(QKV[base + 256 + lane]);
        float vd = QKV[base + 512 + lane];
        kc += kd;
        float z = qd * kc;
        #pragma unroll
        for (int m = 16; m >= 1; m >>= 1) z += __shfl_xor(z, m, 32);
        float num = 0.f;
        #pragma unroll
        for (int d = 0; d < 32; ++d) {
            float kb = __shfl(kd, d, 32);
            float qb = __shfl(qd, d, 32);
            s[d] = fmaf(kb, vd, s[d]);
            num  = fmaf(qb, s[d], num);
        }
        O[((size_t)(b * N_TOT + n)) * 256 + h * D_HEAD_ + lane] = (__bf16)(num / (z + 1e-6f));
    }
}

// ---------------- fused residual-add + LayerNorm (wave per row, D=256) ----------------
// Writes f32 master X and bf16 shadow Xbf (GEMM A operand).
__global__ __launch_bounds__(256) void add_ln_kernel(float* __restrict__ X, const float* __restrict__ Y,
                                                     const float* __restrict__ w, const float* __restrict__ b,
                                                     __bf16* __restrict__ Xbf, int M) {
    const int lane = threadIdx.x & 31;
    const int wid  = threadIdx.x >> 5;
    const int row  = blockIdx.x * 8 + wid;
    if (row >= M) return;
    const size_t base = (size_t)row * 256;
    float v[8];
    float s = 0.f, s2 = 0.f;
    #pragma unroll
    for (int j = 0; j < 8; ++j) {
        int c = lane + 32 * j;
        float t = X[base + c] + Y[base + c];
        v[j] = t; s += t; s2 = fmaf(t, t, s2);
    }
    #pragma unroll
    for (int m = 16; m >= 1; m >>= 1) {
        s  += __shfl_xor(s, m, 32);
        s2 += __shfl_xor(s2, m, 32);
    }
    float mean = s * (1.f / 256.f);
    float var  = s2 * (1.f / 256.f) - mean * mean;
    float inv  = rsqrtf(var + 1e-5f);
    #pragma unroll
    for (int j = 0; j < 8; ++j) {
        int c = lane + 32 * j;
        float o = (v[j] - mean) * inv * w[c] + b[c];
        X[base + c]   = o;
        Xbf[base + c] = (__bf16)o;
    }
}

// ---------------- output heads ----------------
__global__ void heads_kernel(const float* __restrict__ X, const float* __restrict__ meanW,
                             const float* __restrict__ meanb, const float* __restrict__ stdW,
                             const float* __restrict__ stdb, float* __restrict__ out) {
    int tid = blockIdx.x * blockDim.x + threadIdx.x;        // B*S_FUT*16
    int row = tid >> 4;
    int r   = tid & 15;
    int j   = r & 7;
    int is_std = r >> 3;
    int b = row / S_FUT_;
    int t = row - b * S_FUT_;
    const float* xr = X + ((size_t)(b * N_TOT + S_PAST_ + t)) * 256;
    const float* Wp = is_std ? stdW : meanW;
    float acc = is_std ? stdb[j] : meanb[j];
    for (int k = 0; k < 256; ++k) acc = fmaf(xr[k], Wp[k * 8 + j], acc);
    if (is_std) {
        float sp = fmaxf(acc, 0.f) + log1pf(__expf(-fabsf(acc)));
        out[(size_t)B_SZ * S_FUT_ * 8 + row * 8 + j] = 0.01f + 0.99f * sp;
    } else {
        out[row * 8 + j] = acc;
    }
}

// ---------------- orchestration ----------------
extern "C" void kernel_launch(void* const* d_in, const int* in_sizes, int n_in,
                              void* d_out, int out_size, void* d_ws, size_t ws_size,
                              hipStream_t stream) {
    (void)in_sizes; (void)n_in; (void)out_size; (void)ws_size;

    const float* past_x   = (const float*)d_in[0];
    const float* past_y   = (const float*)d_in[1];
    const float* future_x = (const float*)d_in[2];
    const float* enc_W    = (const float*)d_in[3];
    const float* enc_b    = (const float*)d_in[4];
    const float* Wq       = (const float*)d_in[5];
    const float* bq       = (const float*)d_in[6];
    const float* Wk       = (const float*)d_in[7];
    const float* bk       = (const float*)d_in[8];
    const float* Wv       = (const float*)d_in[9];
    const float* bv       = (const float*)d_in[10];
    const float* Wo       = (const float*)d_in[11];
    const float* bo       = (const float*)d_in[12];
    const float* ln1_w    = (const float*)d_in[13];
    const float* ln1_b    = (const float*)d_in[14];
    const float* W1       = (const float*)d_in[15];
    const float* b1       = (const float*)d_in[16];
    const float* W2       = (const float*)d_in[17];
    const float* b2       = (const float*)d_in[18];
    const float* ln2_w    = (const float*)d_in[19];
    const float* ln2_b    = (const float*)d_in[20];
    const float* mean_W   = (const float*)d_in[21];
    const float* mean_b   = (const float*)d_in[22];
    const float* std_W    = (const float*)d_in[23];
    const float* std_b    = (const float*)d_in[24];

    char* ws = (char*)d_ws;
    size_t off = 0;
    auto take = [&](size_t bytes) -> char* {
        char* p = ws + off;
        off += (bytes + 255) & ~(size_t)255;
        return p;
    };
    // transposed bf16 weights
    __bf16* qkvt = (__bf16*)take((size_t)N_LAYERS_ * 768 * 256 * 2);   // [L][768][256]
    __bf16* wot  = (__bf16*)take((size_t)N_LAYERS_ * 256 * 256 * 2);   // [L][256][256]
    __bf16* w1t  = (__bf16*)take((size_t)N_LAYERS_ * 2048 * 256 * 2);  // [L][2048][256]
    __bf16* w2t  = (__bf16*)take((size_t)N_LAYERS_ * 256 * 2048 * 2);  // [L][256][2048]
    float*  bqkv = (float*)take((size_t)N_LAYERS_ * 768 * 4);
    float*  X    = (float*)take((size_t)M_ROWS * 256 * 4);
    __bf16* Xbf  = (__bf16*)take((size_t)M_ROWS * 256 * 2);
    float*  QKV  = (float*)take((size_t)M_ROWS * 768 * 4);
    __bf16* Abf  = (__bf16*)take((size_t)M_ROWS * 256 * 2);
    float*  T1   = (float*)take((size_t)M_ROWS * 256 * 4);
    __bf16* Hf   = (__bf16*)take((size_t)M_ROWS * 2048 * 2);
    float*  chS  = (float*)take((size_t)B_SZ * N_HEADS_ * NCH_ * 32 * 32 * 4);
    float*  chK  = (float*)take((size_t)B_SZ * N_HEADS_ * NCH_ * 32 * 4);

    // 1) weight transforms (f32 [K][N] -> bf16 [N][K]); QKV packed per layer as [768][256]
    cvt_transpose_kernel<<<dim3(8, 8, N_LAYERS_), 256, 0, stream>>>(
        Wq, qkvt,               256, 256, (long)256 * 256, (long)768 * 256);
    cvt_transpose_kernel<<<dim3(8, 8, N_LAYERS_), 256, 0, stream>>>(
        Wk, qkvt + 256 * 256,   256, 256, (long)256 * 256, (long)768 * 256);
    cvt_transpose_kernel<<<dim3(8, 8, N_LAYERS_), 256, 0, stream>>>(
        Wv, qkvt + 512 * 256,   256, 256, (long)256 * 256, (long)768 * 256);
    cvt_transpose_kernel<<<dim3(8, 8, N_LAYERS_), 256, 0, stream>>>(
        Wo, wot,                256, 256, (long)256 * 256, (long)256 * 256);
    cvt_transpose_kernel<<<dim3(64, 8, N_LAYERS_), 256, 0, stream>>>(
        W1, w1t,                256, 2048, (long)256 * 2048, (long)2048 * 256);
    cvt_transpose_kernel<<<dim3(8, 64, N_LAYERS_), 256, 0, stream>>>(
        W2, w2t,                2048, 256, (long)2048 * 256, (long)256 * 2048);
    concat_bias_kernel<<<(N_LAYERS_ * 768) / 256, 256, 0, stream>>>(bq, bk, bv, bqkv);

    // 2) encoder
    encode_kernel<<<(M_ROWS * 256) / 256, 256, 0, stream>>>(
        past_x, past_y, future_x, enc_W, enc_b, X, Xbf);

    const dim3 gQKV(768 / 64,  M_ROWS / 128);
    const dim3 gD  (256 / 64,  M_ROWS / 128);
    const dim3 gFF (D_FF_ / 64, M_ROWS / 128);

    // 3) transformer layers
    for (int i = 0; i < N_LAYERS_; ++i) {
        gemm_bf16_kernel<<<gQKV, 256, 0, stream>>>(Xbf, qkvt + (size_t)i * 768 * 256,
                                                   bqkv + i * 768, QKV, M_ROWS, 768, 256, 0, 0);

        attn_partial_kernel<<<B_SZ * N_HEADS_ * NCH_, 32, 0, stream>>>(QKV, chS, chK);
        attn_prefix_kernel<<<B_SZ * N_HEADS_, 32, 0, stream>>>(chS, chK);
        attn_out_kernel<<<B_SZ * N_HEADS_ * NCH_, 32, 0, stream>>>(QKV, chS, chK, Abf);

        gemm_bf16_kernel<<<gD, 256, 0, stream>>>(Abf, wot + (size_t)i * 256 * 256,
                                                 bo + i * 256, T1, M_ROWS, 256, 256, 0, 0);
        add_ln_kernel<<<M_ROWS / 8, 256, 0, stream>>>(X, T1, ln1_w + i * 256, ln1_b + i * 256,
                                                      Xbf, M_ROWS);

        gemm_bf16_kernel<<<gFF, 256, 0, stream>>>(Xbf, w1t + (size_t)i * 2048 * 256,
                                                  b1 + i * 2048, Hf, M_ROWS, 2048, 256, 1, 1);
        gemm_bf16_kernel<<<gD, 256, 0, stream>>>(Hf, w2t + (size_t)i * 256 * 2048,
                                                 b2 + i * 256, T1, M_ROWS, 256, 2048, 0, 0);
        add_ln_kernel<<<M_ROWS / 8, 256, 0, stream>>>(X, T1, ln2_w + i * 256, ln2_b + i * 256,
                                                      Xbf, M_ROWS);
    }

    // 4) output heads -> d_out = [mean | sigma]
    heads_kernel<<<(B_SZ * S_FUT_ * 16) / 256, 256, 0, stream>>>(
        X, mean_W, mean_b, std_W, std_b, (float*)d_out);
}
